// MultiheadAttention_1417339208425
// MI455X (gfx1250) — matrix operand order repaired and checked by reading, not compiled, not run on hardware
//
#include <hip/hip_runtime.h>
#include <hip/hip_bf16.h>
#include <cstdint>

// Problem constants (from the reference)
#define BB   2
#define SS   2048
#define EE   512
#define HH   8
#define HD   64
#define E3   1536   // 3*E

typedef _Float16 v16h __attribute__((ext_vector_type(16)));
typedef _Float16 v8h  __attribute__((ext_vector_type(8)));
typedef float    v8f  __attribute__((ext_vector_type(8)));

// ---------------------------------------------------------------------------
// WMMA helpers (CDNA5: D = A(16x32 f16) * B(32x16 f16) + C(16x16 f32))
// ---------------------------------------------------------------------------
__device__ __forceinline__ v8f wmma_f16(v16h a, v16h b, v8f c) {
    return __builtin_amdgcn_wmma_f32_16x16x32_f16(
        /*neg_a=*/false, a, /*neg_b=*/false, b,
        /*c_mod=*/(short)0, c, /*reuse_a=*/false, /*reuse_b=*/false);
}

// A fragment (16x32, 16-bit): lane l holds row m=l%16.
// lanes 0-15:  halves 0..15 = K 0..15
// lanes 16-31: halves 0..15 = K 16..31
// rowptr points at (row m, K=0) of the 32-wide K chunk; klo = (l&16)?8:0
__device__ __forceinline__ v16h load_a_frag(const _Float16* rowptr, int klo) {
    v16h a;
    v8h lo = *(const v8h*)(rowptr + 2 * klo);
    v8h hi = *(const v8h*)(rowptr + 2 * klo + 8);
#pragma unroll
    for (int i = 0; i < 8; ++i) { a[i] = lo[i]; a[8 + i] = hi[i]; }
    return a;
}

// B fragment (32x16, 16-bit): lane l holds column n=l%16, K-group kb=(l/16)*16,
// halves e = Bt[n][kb+e]  (Bt = B transposed: N-major storage, contiguous K)
__device__ __forceinline__ v16h load_b_frag(const _Float16* bt_row_plus_kb) {
    return *(const v16h*)(bt_row_plus_kb);
}

// ---------------------------------------------------------------------------
// Prep kernels: f32 -> f16 conversion / transpose
// ---------------------------------------------------------------------------
__global__ void cvt_f32_to_f16(const float* __restrict__ src,
                               _Float16* __restrict__ dst, int n) {
    int i = blockIdx.x * blockDim.x + threadIdx.x;
    if (i < n) dst[i] = (_Float16)src[i];
}

// W[k, n] (rows=K, cols=N, row-major) -> Wt[n, k] f16
__global__ void transpose_cvt(const float* __restrict__ W,
                              _Float16* __restrict__ Wt, int rowsK, int colsN) {
    int i = blockIdx.x * blockDim.x + threadIdx.x;
    int total = rowsK * colsN;
    if (i < total) {
        int n = i / rowsK;
        int k = i - n * rowsK;
        Wt[(size_t)n * rowsK + k] = (_Float16)W[(size_t)k * colsN + n];
    }
}

// ---------------------------------------------------------------------------
// Kernel 1: QKV = X @ Wqkv + bqkv ; scatter into Qh[B,H,S,HD], Kh[B,H,S,HD],
//           Vt[B,H,HD,S]  (all f16).  M=B*S=4096, N=3E=1536, K=E=512.
// One wave computes a 16x64 output strip: A fragment reused across 4 B tiles
// (4 WMMAs per K-step). 64-wide strips never straddle a q/k/v section since
// sections are 64 wide and heads are 192 (=3*64) wide.
// ---------------------------------------------------------------------------
__global__ void qkv_gemm(const _Float16* __restrict__ Xh,
                         const _Float16* __restrict__ WqkvT,
                         const float* __restrict__ bqkv,
                         _Float16* __restrict__ Qh,
                         _Float16* __restrict__ Kh,
                         _Float16* __restrict__ Vt) {
    const int NT64 = E3 / 64;                     // 24 strips of width 64
    int wave = threadIdx.x >> 5;
    int lane = threadIdx.x & 31;
    int tile = blockIdx.x * 8 + wave;             // 256*24 = 6144 wave-tiles
    int mt   = tile / NT64;
    int nt64 = tile - mt * NT64;

    int mrow = mt * 16 + (lane & 15);
    int klo  = (lane & 16) ? 8 : 0;
    int kb   = (lane & 16) ? 16 : 0;

    const _Float16* arow = Xh + (size_t)mrow * EE;
    const _Float16* brow0 =
        WqkvT + (size_t)(nt64 * 64 + (lane & 15)) * EE + kb;

    v8f c0 = {}, c1 = {}, c2 = {}, c3 = {};
    for (int kk = 0; kk < EE; kk += 32) {
        v16h a = load_a_frag(arow + kk, klo);
        c0 = wmma_f16(a, load_b_frag(brow0 + kk), c0);
        c1 = wmma_f16(a, load_b_frag(brow0 + (size_t)16 * EE + kk), c1);
        c2 = wmma_f16(a, load_b_frag(brow0 + (size_t)32 * EE + kk), c2);
        c3 = wmma_f16(a, load_b_frag(brow0 + (size_t)48 * EE + kk), c3);
    }

    v8f acc[4] = {c0, c1, c2, c3};
#pragma unroll
    for (int t = 0; t < 4; ++t) {
        int colg = nt64 * 64 + t * 16 + (lane & 15);
        float bias = bqkv[colg];
        int h  = colg / (3 * HD);
        int j  = colg - h * (3 * HD);
        int sect = j / HD;                        // 0=q 1=k 2=v
        int hd   = j - sect * HD;
#pragma unroll
        for (int r = 0; r < 8; ++r) {
            int rowg = mt * 16 + r + ((lane & 16) ? 8 : 0);
            int bidx = rowg / SS;
            int s    = rowg - bidx * SS;
            _Float16 v = (_Float16)(acc[t][r] + bias);
            size_t bh = (size_t)bidx * HH + h;
            if (sect == 0)      Qh[(bh * SS + s) * HD + hd] = v;
            else if (sect == 1) Kh[(bh * SS + s) * HD + hd] = v;
            else                Vt[(bh * HD + hd) * SS + s] = v;
        }
    }
}

// ---------------------------------------------------------------------------
// Kernel 2: fused attention for one (b,h,16-query-row) block.
//   LDS: sP  16x2048 f32 (128KB)  score strip (reused as f32 partial scratch)
//        sPh 16x2048 f16 ( 64KB)  softmaxed probs for the P@V WMMAs
//        sred/sinv                small reduction scratch
// Phase 1: logits = (Q Kt)/8 via WMMA (8 waves split the 128 key tiles;
//          Q fragments loaded once and reused across the whole loop)
// Phase 2: row softmax in LDS; single coalesced f32 write of attention block
// Phase 3: values = P @ V via WMMA; all 8 waves: 4 HD-tiles x 2 K-halves,
//          halves reduced through LDS
// ---------------------------------------------------------------------------
__global__ void attn_fused(const _Float16* __restrict__ Qh,
                           const _Float16* __restrict__ Kh,
                           const _Float16* __restrict__ Vt,
                           float* __restrict__ attnOut,
                           _Float16* __restrict__ Vals) {
    extern __shared__ char smem[];
    float*    sP   = (float*)smem;                              // 16*2048 f32
    _Float16* sPh  = (_Float16*)(smem + 16 * SS * 4);           // 16*2048 f16
    float*    sred = (float*)(smem + 16 * SS * 4 + 16 * SS * 2);// 16*16 f32
    float*    sinv = sred + 256;                                // 16 f32

    int tid  = threadIdx.x;
    int wave = tid >> 5;
    int lane = tid & 31;

    int qt = blockIdx.x & 127;        // query tile (S/16 = 128)
    int bh = blockIdx.x >> 7;         // b*H + h
    int b  = bh >> 3;
    int h  = bh & 7;

    int mloc = lane & 15;
    int klo  = (lane & 16) ? 8 : 0;
    int kb   = (lane & 16) ? 16 : 0;

    // --- Phase 1: logits ----------------------------------------------------
    const _Float16* qrow = Qh + ((size_t)bh * SS + qt * 16 + mloc) * HD;
    v16h a0 = load_a_frag(qrow, klo);          // K-dim 0..31 of HD
    v16h a1 = load_a_frag(qrow + 32, klo);     // K-dim 32..63

    for (int kt = wave; kt < SS / 16; kt += 8) {
        const _Float16* krow = Kh + ((size_t)bh * SS + kt * 16 + mloc) * HD + kb;
        v16h b0 = load_b_frag(krow);
        v16h b1 = load_b_frag(krow + 32);
        v8f c = {};
        c = wmma_f16(a0, b0, c);
        c = wmma_f16(a1, b1, c);
#pragma unroll
        for (int r = 0; r < 8; ++r) {
            int row = r + ((lane & 16) ? 8 : 0);
            sP[row * SS + kt * 16 + (lane & 15)] = c[r] * 0.125f; // 1/sqrt(64)
        }
    }
    __syncthreads();

    // --- Phase 2: softmax over each of the 16 rows (16 threads per row) ----
    int row = tid >> 4;
    int p   = tid & 15;

    float mx = -3.402823e38f;
    for (int c2 = p; c2 < SS; c2 += 16) mx = fmaxf(mx, sP[row * SS + c2]);
    sred[row * 16 + p] = mx;
    __syncthreads();
    for (int off = 8; off > 0; off >>= 1) {
        if (p < off)
            sred[row * 16 + p] = fmaxf(sred[row * 16 + p], sred[row * 16 + p + off]);
        __syncthreads();
    }
    mx = sred[row * 16];
    __syncthreads();

    float sum = 0.f;
    for (int c2 = p; c2 < SS; c2 += 16) {
        float e = __expf(sP[row * SS + c2] - mx);
        sP[row * SS + c2] = e;
        sum += e;
    }
    sred[row * 16 + p] = sum;
    __syncthreads();
    for (int off = 8; off > 0; off >>= 1) {
        if (p < off)
            sred[row * 16 + p] += sred[row * 16 + p + off];
        __syncthreads();
    }
    if (p == 0) sinv[row] = 1.0f / sred[row * 16];
    __syncthreads();

    // Coalesced normalize + write attention block + stash f16 probs
    float* aout = attnOut + ((size_t)bh * SS + qt * 16) * SS;
    for (int idx = tid; idx < 16 * SS; idx += 256) {
        int ri = idx >> 11;            // /2048
        float pv = sP[idx] * sinv[ri];
        aout[(size_t)ri * SS + (idx & (SS - 1))] = pv;
        sPh[idx] = (_Float16)pv;
    }
    __syncthreads();

    // --- Phase 3: values = P @ V; all 8 waves (4 HD-tiles x 2 K-halves) ----
    {
        int nt    = wave & 3;          // HD tile (16 wide)
        int khalf = wave >> 2;         // which half of S (1024 keys)
        const _Float16* vbase =
            Vt + ((size_t)bh * HD + nt * 16 + (lane & 15)) * SS
               + khalf * (SS / 2) + kb;
        const _Float16* pbase = sPh + mloc * SS + khalf * (SS / 2);
        v8f c = {};
        for (int kt = 0; kt < SS / 64; ++kt) {   // 32 chunks of 32 keys
            v16h a   = load_a_frag(pbase + kt * 32, klo);
            v16h bfr = load_b_frag(vbase + kt * 32);
            c = wmma_f16(a, bfr, c);
        }
        // partial results -> LDS scratch (sP region is free now)
        // layout: scratch[khalf][row 0..15][hd 0..63]
#pragma unroll
        for (int r = 0; r < 8; ++r) {
            int rr = r + ((lane & 16) ? 8 : 0);
            sP[(khalf * 16 + rr) * 64 + nt * 16 + (lane & 15)] = c[r];
        }
    }
    __syncthreads();

    // Reduce the two K-halves and write values (head-interleaved, f16)
    for (int idx = tid; idx < 16 * 64; idx += 256) {
        int ri = idx >> 6;
        int cd = idx & 63;
        float v = sP[ri * 64 + cd] + sP[(16 + ri) * 64 + cd];
        int srow = qt * 16 + ri;
        Vals[((size_t)b * SS + srow) * EE + h * HD + cd] = (_Float16)v;
    }
}

// ---------------------------------------------------------------------------
// Kernel 3: O = Vals @ Wo + bo   (M=4096, N=512, K=512), f32 output.
// 16x64 strip per wave, A fragment reused across 4 B tiles.
// ---------------------------------------------------------------------------
__global__ void out_gemm(const _Float16* __restrict__ Vals,
                         const _Float16* __restrict__ WoT,
                         const float* __restrict__ bo,
                         float* __restrict__ O) {
    const int NT64 = EE / 64;                     // 8 strips
    int wave = threadIdx.x >> 5;
    int lane = threadIdx.x & 31;
    int tile = blockIdx.x * 8 + wave;             // 256*8 = 2048 wave-tiles
    int mt   = tile / NT64;
    int nt64 = tile - mt * NT64;

    int mrow = mt * 16 + (lane & 15);
    int klo  = (lane & 16) ? 8 : 0;
    int kb   = (lane & 16) ? 16 : 0;

    const _Float16* arow = Vals + (size_t)mrow * EE;
    const _Float16* brow0 =
        WoT + (size_t)(nt64 * 64 + (lane & 15)) * EE + kb;

    v8f c0 = {}, c1 = {}, c2 = {}, c3 = {};
    for (int kk = 0; kk < EE; kk += 32) {
        v16h a = load_a_frag(arow + kk, klo);
        c0 = wmma_f16(a, load_b_frag(brow0 + kk), c0);
        c1 = wmma_f16(a, load_b_frag(brow0 + (size_t)16 * EE + kk), c1);
        c2 = wmma_f16(a, load_b_frag(brow0 + (size_t)32 * EE + kk), c2);
        c3 = wmma_f16(a, load_b_frag(brow0 + (size_t)48 * EE + kk), c3);
    }

    v8f acc[4] = {c0, c1, c2, c3};
#pragma unroll
    for (int t = 0; t < 4; ++t) {
        int colg = nt64 * 64 + t * 16 + (lane & 15);
        float bias = bo[colg];
#pragma unroll
        for (int r = 0; r < 8; ++r) {
            int rowg = mt * 16 + r + ((lane & 16) ? 8 : 0);
            O[(size_t)rowg * EE + colg] = acc[t][r] + bias;
        }
    }
}

// ---------------------------------------------------------------------------
// Host launch
// ---------------------------------------------------------------------------
extern "C" void kernel_launch(void* const* d_in, const int* in_sizes, int n_in,
                              void* d_out, int out_size, void* d_ws, size_t ws_size,
                              hipStream_t stream) {
    (void)in_sizes; (void)n_in; (void)out_size; (void)ws_size;

    const float* X    = (const float*)d_in[0];   // [B,S,E]
    const float* Wqkv = (const float*)d_in[1];   // [E,3E]
    const float* bqkv = (const float*)d_in[2];   // [3E]
    const float* Wo   = (const float*)d_in[3];   // [E,E]
    const float* bo   = (const float*)d_in[4];   // [E]

    float* o_out    = (float*)d_out;                               // [B,S,E]
    float* attn_out = o_out + (size_t)BB * SS * EE;                // [B,H,S,S]

    // Workspace carve-up (256B aligned slabs), 22 MB total
    char*  ws  = (char*)d_ws;
    size_t off = 0;
    auto carve = [&](size_t bytes) {
        char* p = ws + off;
        off += (bytes + 255) & ~(size_t)255;
        return p;
    };
    _Float16* Xh     = (_Float16*)carve((size_t)BB * SS * EE * 2);       // 4 MB
    _Float16* WqkvT  = (_Float16*)carve((size_t)E3 * EE * 2);            // 1.5 MB
    _Float16* WoT    = (_Float16*)carve((size_t)EE * EE * 2);            // 0.5 MB
    _Float16* Qh     = (_Float16*)carve((size_t)BB * HH * SS * HD * 2);  // 4 MB
    _Float16* Kh     = (_Float16*)carve((size_t)BB * HH * SS * HD * 2);  // 4 MB
    _Float16* Vt     = (_Float16*)carve((size_t)BB * HH * HD * SS * 2);  // 4 MB
    _Float16* Vals   = (_Float16*)carve((size_t)BB * SS * EE * 2);       // 4 MB

    // 1) precision prep
    {
        int n = BB * SS * EE;
        cvt_f32_to_f16<<<(n + 255) / 256, 256, 0, stream>>>(X, Xh, n);
    }
    transpose_cvt<<<(EE * E3 + 255) / 256, 256, 0, stream>>>(Wqkv, WqkvT, EE, E3);
    transpose_cvt<<<(EE * EE + 255) / 256, 256, 0, stream>>>(Wo, WoT, EE, EE);

    // 2) QKV projection: 256 M-tiles * 24 N64-strips / 8 waves = 768 blocks
    qkv_gemm<<<768, 256, 0, stream>>>(Xh, WqkvT, bqkv, Qh, Kh, Vt);

    // 3) fused attention: B*H*(S/16) = 2048 blocks, ~193 KB dynamic LDS
    size_t smem = (size_t)16 * SS * 4 + (size_t)16 * SS * 2 + 256 * 4 + 16 * 4;
    (void)hipFuncSetAttribute((const void*)attn_fused,
                              hipFuncAttributeMaxDynamicSharedMemorySize,
                              (int)smem);
    attn_fused<<<BB * HH * (SS / 16), 256, smem, stream>>>(Qh, Kh, Vt,
                                                           attn_out, Vals);

    // 4) output projection: 256 M-tiles * 8 N64-strips / 8 waves = 256 blocks
    out_gemm<<<256, 256, 0, stream>>>(Vals, WoT, bo, o_out);
}
